// SwinBlock_39865886441839
// MI455X (gfx1250) — matrix operand order repaired and checked by reading, not compiled, not run on hardware
//
#include <hip/hip_runtime.h>
#include <hip/hip_bf16.h>

// ---------------------------------------------------------------------------
// Swin block for MI455X (gfx1250, wave32, WMMA 16x16x32 f16).
// GEMMs: v_wmma_f32_16x16x32_f16, f16 activations, fp32 accumulate.
// Global->LDS staging in the GEMMs uses the Tensor Data Mover
// (tensor_load_to_lds, TENSORcnt) with double-buffered LDS, hardware row
// padding reproducing the 16B-aligned fragment-friendly LDS layouts.
// B operands kept transposed ([N][K]) so A and B fragments are 2x
// ds_load_b128 per lane.
// ---------------------------------------------------------------------------

typedef __attribute__((ext_vector_type(16))) _Float16 v16h;
typedef __attribute__((ext_vector_type(8)))  _Float16 h8;
typedef __attribute__((ext_vector_type(8)))  float    v8f;
typedef unsigned int u32x4 __attribute__((ext_vector_type(4)));
typedef int          i32x8 __attribute__((ext_vector_type(8)));
typedef int          i32x4 __attribute__((ext_vector_type(4)));

#define DIM   384
#define NTOK  65536     // 16*64*64 tokens
#define QKVN  1152
#define MLPN  1536

__device__ inline v8f wmma32(v16h a, v16h b, v8f c) {
  return __builtin_amdgcn_wmma_f32_16x16x32_f16(
      /*neg_a=*/false, a, /*neg_b=*/false, b,
      /*c_mod=*/(short)0, c, /*reuse_a=*/false, /*reuse_b=*/false);
}

// A fragment (16x32 MxK f16), row-major [M][K], ld in halves (mult of 8).
__device__ inline v16h frag_a(const _Float16* __restrict__ s, int ld) {
  int lane = threadIdx.x & 31;
  int m  = lane & 15;
  int kb = (lane >> 4) << 3;
  const _Float16* r = s + m * ld + kb;
  v16h a;
  *(h8*)&a       = *(const h8*)(r);
  *((h8*)&a + 1) = *(const h8*)(r + 16);
  return a;
}

// B fragment (32x16 KxN f16) from TRANSPOSED source: s = B^T row-major
// [N][K], ld in halves (mult of 8). b[j] = s[n*ld+kb+j], b[8+j] = s[n*ld+kb+16+j].
__device__ inline v16h frag_bT(const _Float16* __restrict__ s, int ld) {
  int lane = threadIdx.x & 31;
  int n  = lane & 15;
  int kb = (lane >> 4) << 3;
  const _Float16* r = s + n * ld + kb;
  v16h b;
  *(h8*)&b       = *(const h8*)(r);
  *((h8*)&b + 1) = *(const h8*)(r + 16);
  return b;
}

// ---------------------------------------------------------------------------
// TDM: 2D tile load Global->LDS. data_size=2B. Row padding in LDS via
// pad_interval/pad_amount (D# group1). Groups 2/3 (+extra group) zero for a
// 2-D tensor. D# packing per CDNA5 ISA 8.3/8.4. 6-arg builtin (clang-23).
// ---------------------------------------------------------------------------
__device__ inline void tdm_load_2d(void* lds_dst, const void* gsrc,
                                   unsigned tensor_d0, unsigned tensor_d1,
                                   unsigned tile_d0,   unsigned tile_d1,
                                   unsigned d0_stride,
                                   unsigned pad_interval, unsigned pad_amount) {
  unsigned long long ga = (unsigned long long)(uintptr_t)gsrc;
  u32x4 g0;
  g0[0] = 1u;                                    // count=1, user descriptor
  g0[1] = (unsigned)(uintptr_t)lds_dst;          // lds_addr (bytes)
  g0[2] = (unsigned)ga;                          // global_addr[31:0]
  g0[3] = (unsigned)((ga >> 32) & 0x01ffffffu)   // global_addr[56:32]
        | (2u << 30);                            // type=2 ("image")
  i32x8 g1;
  g1[0] = (int)((1u << 16)                       // data_size: 2 bytes
              | (1u << 20)                       // pad_enable
              | (pad_interval << 22)
              | (pad_amount << 25));
  g1[1] = (int)((tensor_d0 & 0xffffu) << 16);          // tensor_dim0[15:0]
  g1[2] = (int)((tensor_d0 >> 16)                      // tensor_dim0[31:16]
              | ((tensor_d1 & 0xffffu) << 16));        // tensor_dim1[15:0]
  g1[3] = (int)((tensor_d1 >> 16) | (tile_d0 << 16));  // dim1 hi | tile_dim0
  g1[4] = (int)tile_d1;                                // tile_dim1 (dim2=0)
  g1[5] = (int)d0_stride;                              // dim0_stride[31:0]
  g1[6] = 0;                                           // stride hi / dim1_stride
  g1[7] = 0;
  i32x4 z4 = {0, 0, 0, 0};
  i32x8 z8 = {0, 0, 0, 0, 0, 0, 0, 0};
  __builtin_amdgcn_tensor_load_to_lds(g0, g1, z4, z4, z8, 0);
}

// ---------------------------------------------------------------------------
// K0: fp32 [K][N] -> f16 transposed [N][K] weight conversion (LDS tiled).
// ---------------------------------------------------------------------------
__global__ __launch_bounds__(256)
void convT_f16(const float* __restrict__ s, _Float16* __restrict__ d,
               int K, int N) {
  __shared__ _Float16 t[32][33];
  int k0 = blockIdx.y * 32, n0 = blockIdx.x * 32;
  int c  = threadIdx.x & 31;
  int r0 = threadIdx.x >> 5;
  for (int r = r0; r < 32; r += 8)
    t[r][c] = (_Float16)s[(size_t)(k0 + r) * N + n0 + c];
  __syncthreads();
  for (int r = r0; r < 32; r += 8)
    d[(size_t)(n0 + r) * K + k0 + c] = t[c][r];
}

// ---------------------------------------------------------------------------
// K1/K5: RMSNorm over channels + gather into GEMM row order (f16).
// ---------------------------------------------------------------------------
template<bool SHIFTED>
__global__ __launch_bounds__(256)
void rmsnorm_gather(const float* __restrict__ src, const float* __restrict__ w,
                    _Float16* __restrict__ out) {
  __shared__ float ss[256];
  __shared__ float inv[64];
  __shared__ _Float16 tile[64 * DIM];   // 48 KB

  int b  = blockIdx.x >> 6;
  int hh = blockIdx.x & 63;
  int t  = threadIdx.x;
  int wc = t & 63;
  int cs = t >> 6;

  const float* base = src + (((size_t)b * DIM) * 64 + hh) * 64;

  float acc = 0.f;
  for (int c = cs; c < DIM; c += 4) {
    float v = base[(size_t)c * 4096 + wc];
    acc += v * v;
  }
  ss[t] = acc;
  __syncthreads();
  if (t < 64) {
    float s = ss[t] + ss[t + 64] + ss[t + 128] + ss[t + 192];
    inv[t] = rsqrtf(s * (1.0f / DIM) + 1e-6f);
  }
  __syncthreads();
  float iv = inv[wc];
  for (int c = cs; c < DIM; c += 4) {
    float v = base[(size_t)c * 4096 + wc] * iv * w[c];
    tile[wc * DIM + c] = (_Float16)v;
  }
  __syncthreads();

  int hr = SHIFTED ? ((hh + 60) & 63) : hh;
#pragma unroll
  for (int n = 0; n < 12; ++n) {
    int id = t + n * 256;
    int r = id / 48;
    int q = id % 48;
    size_t row;
    if (SHIFTED) {
      int wr = (r + 60) & 63;
      row = (((size_t)b * 64 + (hr >> 3) * 8 + (wr >> 3)) * 64) +
            ((hr & 7) * 8 + (wr & 7));
    } else {
      row = ((size_t)b * 64 + hh) * 64 + r;
    }
    *(h8*)(out + row * DIM + q * 8) = *(const h8*)(tile + r * DIM + q * 8);
  }
}

// ---------------------------------------------------------------------------
// GEMM: block tile 128(M) x 64(N), K-step 32, 8 waves (4x2), each wave a
// 32x32 quad. TDM stages A [128x32 -> ld48] and B^T [64x32 -> ld40] into
// double-buffered LDS, pipelined one K-tile ahead; fragments are 2x
// ds_load_b128. Bt argument is pre-transposed weights [N][K].
// MODE 0: out f16 = acc + bias                        (QKV)
// MODE 1: out f16 = gelu_exact(acc + bias)            (MLP1)
// MODE 2: out f32[unshift-scatter] = x + acc + bias   (proj + shortcut)
// MODE 3: out f32[raster-scatter]  = res + acc + bias (MLP2 + residual)
// ---------------------------------------------------------------------------
template<int MODE>
__global__ __launch_bounds__(256)
void gemm_wmma(const _Float16* __restrict__ A, const _Float16* __restrict__ Bt,
               const float* __restrict__ bias, int M, int N, int K,
               _Float16* __restrict__ outh, float* __restrict__ outf,
               const float* __restrict__ addsrc) {
  __shared__ _Float16 As[2][128 * 48];   // 12 KB each
  __shared__ _Float16 Bs[2][64 * 40];    //  5 KB each

  int bn = blockIdx.x * 64;
  int bm = blockIdx.y * 128;
  int t  = threadIdx.x;
  int wv = t >> 5;
  int lane = t & 31;
  int mw = (wv & 3) * 32;
  int nw = (wv >> 2) * 32;

  const int nk = K >> 5;
  v8f acc[2][2] = {};

  // prologue: K-tile 0 -> buffer 0 (wave 0 drives the TDM)
  if (wv == 0) {
    tdm_load_2d(As[0], A  + (size_t)bm * K, K, M, 32, 128, K, 3u, 7u);
    tdm_load_2d(Bs[0], Bt + (size_t)bn * K, K, N, 32, 64,  K, 3u, 3u);
  }

  for (int kt = 0; kt < nk; ++kt) {
    int cur = kt & 1;
    if (wv == 0) {
      if (kt + 1 < nk) {
        int k1 = (kt + 1) << 5;
        tdm_load_2d(As[cur ^ 1], A  + (size_t)bm * K + k1, K, M, 32, 128, K, 3u, 7u);
        tdm_load_2d(Bs[cur ^ 1], Bt + (size_t)bn * K + k1, K, N, 32, 64,  K, 3u, 3u);
        __builtin_amdgcn_s_wait_tensorcnt(2);   // tile kt complete (in-order)
      } else {
        __builtin_amdgcn_s_wait_tensorcnt(0);
      }
    }
    __syncthreads();

    v16h a0 = frag_a (As[cur] + (mw)      * 48, 48);
    v16h a1 = frag_a (As[cur] + (mw + 16) * 48, 48);
    v16h b0 = frag_bT(Bs[cur] + (nw)      * 40, 40);
    v16h b1 = frag_bT(Bs[cur] + (nw + 16) * 40, 40);
    acc[0][0] = wmma32(a0, b0, acc[0][0]);
    acc[0][1] = wmma32(a0, b1, acc[0][1]);
    acc[1][0] = wmma32(a1, b0, acc[1][0]);
    acc[1][1] = wmma32(a1, b1, acc[1][1]);
    __syncthreads();
  }

  int ncol = lane & 15;
  int half = lane >> 4;
#pragma unroll
  for (int i = 0; i < 2; ++i) {
#pragma unroll
    for (int j = 0; j < 2; ++j) {
      int col0 = bn + nw + j * 16;
      int row0 = bm + mw + i * 16;
      int c = col0 + ncol;
      float bi = bias[c];
#pragma unroll
      for (int vg = 0; vg < 8; ++vg) {
        int m = row0 + vg + half * 8;    // C-layout: lanes16-31 hold M=vg+8
        float val = acc[i][j][vg] + bi;
        if (MODE == 0) {
          outh[(size_t)m * N + c] = (_Float16)val;
        } else if (MODE == 1) {
          float g = 0.5f * val * (1.0f + erff(val * 0.70710678118654752f));
          outh[(size_t)m * N + c] = (_Float16)g;
        } else if (MODE == 2) {
          int win = m >> 6, tt = m & 63;
          int b = win >> 6, wi = win & 63;
          int hh = (((wi >> 3) << 3) + (tt >> 3) + 4) & 63;  // un-roll (+4)
          int ww = (((wi & 7) << 3) + (tt & 7) + 4) & 63;
          size_t idx = ((((size_t)b * DIM + c) * 64) + hh) * 64 + ww;
          outf[idx] = addsrc[idx] + val;
        } else {
          int b = m >> 12, hh = (m >> 6) & 63, ww = m & 63;
          size_t idx = ((((size_t)b * DIM + c) * 64) + hh) * 64 + ww;
          outf[idx] = addsrc[idx] + val;
        }
      }
    }
  }
}

// ---------------------------------------------------------------------------
// K3: windowed attention, one wave per (window, head) -> 8192 waves.
// q, k staged natural [token][dim] (k natural == B^T for q@k^T), v staged
// transposed [dim][token] (== B^T for P@v); head_dim 48 zero-padded to 64.
// 32 WMMA scores, fp32 register softmax (16-lane shfl_xor), P restaged via
// LDS, 24 WMMA for P@v.
// ---------------------------------------------------------------------------
__global__ __launch_bounds__(32)
void win_attn(const _Float16* __restrict__ qkv, _Float16* __restrict__ att) {
  __shared__ _Float16 q_s[64 * 64];   // reused for P after softmax
  __shared__ _Float16 k_s[64 * 64];
  __shared__ _Float16 vT [64 * 64];

  int win  = blockIdx.x >> 3;
  int head = blockIdx.x & 7;
  int lane = threadIdx.x;

  const _Float16* base = qkv + (size_t)win * 64 * QKVN + head * 48;

#pragma unroll
  for (int i = 0; i < 12; ++i) {
    int id  = lane + i * 32;
    int row = id / 6;
    int ch  = (id % 6) * 8;
    const _Float16* g = base + (size_t)row * QKVN + ch;
    *(h8*)(q_s + row * 64 + ch) = *(const h8*)(g);
    *(h8*)(k_s + row * 64 + ch) = *(const h8*)(g + DIM);
  }
  h8 zero8 = {};
#pragma unroll
  for (int i = 0; i < 4; ++i) {
    int id  = lane + i * 32;
    int row = id >> 1;
    int ch  = 48 + (id & 1) * 8;
    *(h8*)(q_s + row * 64 + ch) = zero8;
    *(h8*)(k_s + row * 64 + ch) = zero8;
  }
#pragma unroll
  for (int rr = 0; rr < 2; ++rr) {
    int row = lane + rr * 32;
    const _Float16* vr = base + (size_t)row * QKVN + 2 * DIM;
    h8 buf[6];
#pragma unroll
    for (int ch = 0; ch < 6; ++ch) buf[ch] = *(const h8*)(vr + ch * 8);
#pragma unroll
    for (int ch = 0; ch < 6; ++ch)
#pragma unroll
      for (int e = 0; e < 8; ++e)
        vT[(ch * 8 + e) * 64 + row] = buf[ch][e];
#pragma unroll
    for (int dd = 48; dd < 64; ++dd) vT[dd * 64 + row] = (_Float16)0.f;
  }
  __syncthreads();

  v8f s[4][4] = {};
#pragma unroll
  for (int kk = 0; kk < 64; kk += 32) {
    v16h a[4], bf[4];
#pragma unroll
    for (int mi = 0; mi < 4; ++mi) a[mi]  = frag_a (q_s + mi * 16 * 64 + kk, 64);
#pragma unroll
    for (int ni = 0; ni < 4; ++ni) bf[ni] = frag_bT(k_s + ni * 16 * 64 + kk, 64);
#pragma unroll
    for (int mi = 0; mi < 4; ++mi)
#pragma unroll
      for (int ni = 0; ni < 4; ++ni)
        s[mi][ni] = wmma32(a[mi], bf[ni], s[mi][ni]);
  }

  const float scale = 0.14433756729740643f;  // 1/sqrt(48)
  int half = lane >> 4;
  int ncol = lane & 15;
#pragma unroll
  for (int mi = 0; mi < 4; ++mi) {
#pragma unroll
    for (int vg = 0; vg < 8; ++vg) {
      float z0 = s[mi][0][vg] * scale;
      float z1 = s[mi][1][vg] * scale;
      float z2 = s[mi][2][vg] * scale;
      float z3 = s[mi][3][vg] * scale;
      float mx = fmaxf(fmaxf(z0, z1), fmaxf(z2, z3));
#pragma unroll
      for (int off = 1; off < 16; off <<= 1)
        mx = fmaxf(mx, __shfl_xor(mx, off, 32));
      float e0 = __expf(z0 - mx), e1 = __expf(z1 - mx);
      float e2 = __expf(z2 - mx), e3 = __expf(z3 - mx);
      float sum = e0 + e1 + e2 + e3;
#pragma unroll
      for (int off = 1; off < 16; off <<= 1)
        sum += __shfl_xor(sum, off, 32);
      float r = 1.0f / sum;
      int mrow = mi * 16 + vg + half * 8;
      q_s[mrow * 64 + 0  + ncol] = (_Float16)(e0 * r);
      q_s[mrow * 64 + 16 + ncol] = (_Float16)(e1 * r);
      q_s[mrow * 64 + 32 + ncol] = (_Float16)(e2 * r);
      q_s[mrow * 64 + 48 + ncol] = (_Float16)(e3 * r);
    }
  }
  __syncthreads();

  v8f o[4][3] = {};
#pragma unroll
  for (int kk = 0; kk < 64; kk += 32) {
    v16h a[4], bf[3];
#pragma unroll
    for (int mi = 0; mi < 4; ++mi) a[mi]  = frag_a (q_s + mi * 16 * 64 + kk, 64);
#pragma unroll
    for (int ni = 0; ni < 3; ++ni) bf[ni] = frag_bT(vT + ni * 16 * 64 + kk, 64);
#pragma unroll
    for (int mi = 0; mi < 4; ++mi)
#pragma unroll
      for (int ni = 0; ni < 3; ++ni)
        o[mi][ni] = wmma32(a[mi], bf[ni], o[mi][ni]);
  }

  _Float16* ob = att + (size_t)win * 64 * DIM + head * 48;
#pragma unroll
  for (int mi = 0; mi < 4; ++mi)
#pragma unroll
    for (int ni = 0; ni < 3; ++ni)
#pragma unroll
      for (int vg = 0; vg < 8; ++vg) {
        int mrow = mi * 16 + vg + half * 8;
        ob[(size_t)mrow * DIM + ni * 16 + ncol] = (_Float16)o[mi][ni][vg];
      }
}

// ---------------------------------------------------------------------------
// Host-side launch
// ---------------------------------------------------------------------------
extern "C" void kernel_launch(void* const* d_in, const int* in_sizes, int n_in,
                              void* d_out, int out_size, void* d_ws, size_t ws_size,
                              hipStream_t stream) {
  const float* x      = (const float*)d_in[0];
  const float* n1w    = (const float*)d_in[1];
  const float* n2w    = (const float*)d_in[2];
  const float* w_qkv  = (const float*)d_in[3];
  const float* b_qkv  = (const float*)d_in[4];
  const float* w_proj = (const float*)d_in[5];
  const float* b_proj = (const float*)d_in[6];
  const float* w_mlp1 = (const float*)d_in[7];
  const float* b_mlp1 = (const float*)d_in[8];
  const float* w_mlp2 = (const float*)d_in[9];
  const float* b_mlp2 = (const float*)d_in[10];
  float* out = (float*)d_out;

  // Workspace (~340 MB):
  //   [0, 4MB)  f16 transposed weights
  //   res fp32 NCHW       100663296 B
  //   A1  f16 [Ntok,384]   50331648 B  (reused as MLP input)
  //   QKV f16 [Ntok,1152] 150994944 B \ reused together as MLP hidden
  //   ATT f16 [Ntok,384]   50331648 B /  (exactly 201326592 B)
  char* ws = (char*)d_ws;
  _Float16* wqkT = (_Float16*)ws;        // [1152][384]
  _Float16* wprT = wqkT + 442368;        // [384][384]
  _Float16* wm1T = wprT + 147456;        // [1536][384]
  _Float16* wm2T = wm1T + 589824;        // [384][1536]
  float*    res  = (float*)(ws + (4u << 20));
  _Float16* A1   = (_Float16*)((char*)res + 100663296u);
  _Float16* QKV  = (_Float16*)((char*)A1 + 50331648u);
  _Float16* ATT  = (_Float16*)((char*)QKV + 150994944u);
  _Float16* Hb   = QKV;   // 201 MB = QKV+ATT regions (ATT dead by then)

  // K0: weight convert + transpose ([K][N] f32 -> [N][K] f16)
  convT_f16<<<dim3(QKVN / 32, DIM / 32),  256, 0, stream>>>(w_qkv,  wqkT, DIM,  QKVN);
  convT_f16<<<dim3(DIM / 32,  DIM / 32),  256, 0, stream>>>(w_proj, wprT, DIM,  DIM);
  convT_f16<<<dim3(MLPN / 32, DIM / 32),  256, 0, stream>>>(w_mlp1, wm1T, DIM,  MLPN);
  convT_f16<<<dim3(DIM / 32,  MLPN / 32), 256, 0, stream>>>(w_mlp2, wm2T, MLPN, DIM);

  // K1: norm1 + shift + window gather
  rmsnorm_gather<true><<<1024, 256, 0, stream>>>(x, n1w, A1);

  // K2: QKV GEMM  [65536,384] x [384,1152]
  gemm_wmma<0><<<dim3(QKVN / 64, NTOK / 128), 256, 0, stream>>>(
      A1, wqkT, b_qkv, NTOK, QKVN, DIM, QKV, nullptr, nullptr);

  // K3: attention, one wave per (window, head)
  win_attn<<<8192, 32, 0, stream>>>(QKV, ATT);

  // K4: proj GEMM + un-shift scatter + shortcut add -> res (fp32 NCHW)
  gemm_wmma<2><<<dim3(DIM / 64, NTOK / 128), 256, 0, stream>>>(
      ATT, wprT, b_proj, NTOK, DIM, DIM, nullptr, res, x);

  // K5: norm2 gather (raster order)
  rmsnorm_gather<false><<<1024, 256, 0, stream>>>(res, n2w, A1);

  // K6: MLP1 GEMM + exact GELU -> H
  gemm_wmma<1><<<dim3(MLPN / 64, NTOK / 128), 256, 0, stream>>>(
      A1, wm1T, b_mlp1, NTOK, MLPN, DIM, Hb, nullptr, nullptr);

  // K7: MLP2 GEMM + residual -> out (fp32 NCHW)
  gemm_wmma<3><<<dim3(DIM / 64, NTOK / 128), 256, 0, stream>>>(
      Hb, wm2T, b_mlp2, NTOK, DIM, MLPN, nullptr, out, res);
}
